// SpatialConvolutionModule_86620900425892
// MI455X (gfx1250) — matrix operand order repaired and checked by reading, not compiled, not run on hardware
//
#include <hip/hip_runtime.h>
#include <cmath>
#include <cstdint>

typedef __attribute__((ext_vector_type(16))) _Float16 v16h;
typedef __attribute__((ext_vector_type(8)))  _Float16 v8h;
typedef __attribute__((ext_vector_type(8)))  float    v8f;

#define D 128
#define TILE_M 128

// ---------------------------------------------------------------------------
// 1) init min/max buffer
__global__ void init_mm_kernel(int* mm) {
    mm[0] = 0x7f800000;  // +inf bits (min slot, dim0)
    mm[1] = 0x7f800000;  // +inf bits (min slot, dim1)
    mm[2] = 0x80000000;  // INT_MIN   (max slot, dim0)
    mm[3] = 0x80000000;  // INT_MIN   (max slot, dim1)
}

// 2) coordinate min/max via int-punned atomics (coords are non-negative)
__global__ __launch_bounds__(256) void minmax_kernel(const float* __restrict__ coords,
                                                     int* __restrict__ mm, int n2) {
    int i = blockIdx.x * blockDim.x + threadIdx.x;
    if (i >= n2) return;
    int bits = __float_as_int(coords[i]);
    int dim = i & 1;
    atomicMin(&mm[dim], bits);
    atomicMax(&mm[2 + dim], bits);
}

// 3) scatter-add features into grid, record flat index per point
__global__ __launch_bounds__(128) void scatter_kernel(const float* __restrict__ feat,
                                                      const float* __restrict__ coords,
                                                      const int* __restrict__ mm,
                                                      int* __restrict__ flat_out,
                                                      float* __restrict__ sums,
                                                      float* __restrict__ counts,
                                                      int H, int W) {
    int i = blockIdx.x;
    int t = threadIdx.x;
    float c0 = coords[2 * i], c1 = coords[2 * i + 1];
    float mn0 = __int_as_float(mm[0]), mn1 = __int_as_float(mm[1]);
    float mx0 = __int_as_float(mm[2]), mx1 = __int_as_float(mm[3]);
    float g0f = (c0 - mn0) / (mx0 - mn0 + 1e-8f) * (float)(H - 1);
    float g1f = (c1 - mn1) / (mx1 - mn1 + 1e-8f) * (float)(W - 1);
    int g0 = (int)g0f; g0 = min(max(g0, 0), H - 1);
    int g1 = (int)g1f; g1 = min(max(g1, 0), W - 1);
    int fl = g0 * W + g1;
    atomicAdd(&sums[(size_t)fl * D + t], feat[(size_t)i * D + t]);
    if (t == 0) {
        flat_out[i] = fl;
        atomicAdd(&counts[fl], 1.0f);
    }
}

// 4) repack OIHW fp32 weights -> [tap][o][c] fp16 (c contiguous = K contiguous)
__global__ __launch_bounds__(256) void wprep_kernel(const float* __restrict__ w1,
                                                    const float* __restrict__ w2,
                                                    _Float16* __restrict__ o1,
                                                    _Float16* __restrict__ o2, int wtot) {
    int idx = blockIdx.x * blockDim.x + threadIdx.x;
    if (idx >= 2 * wtot) return;
    const float* src = (idx < wtot) ? w1 : w2;
    _Float16*    dst = (idx < wtot) ? o1 : o2;
    int r   = (idx < wtot) ? idx : (idx - wtot);
    int tap = r >> 14;          // / (128*128)
    int rem = r & 16383;
    int o   = rem >> 7;
    int c   = rem & 127;
    dst[r] = (_Float16)src[o * (D * 9) + c * 9 + tap];
}

// 5) grid mean + convert to fp16 channels-last
__global__ __launch_bounds__(256) void grid_norm_kernel(const float* __restrict__ sums,
                                                        const float* __restrict__ counts,
                                                        _Float16* __restrict__ g, size_t n) {
    size_t i = (size_t)blockIdx.x * blockDim.x + threadIdx.x;
    if (i >= n) return;
    float c = counts[i >> 7];
    g[i] = (_Float16)(sums[i] / fmaxf(c, 1.0f));
}

// ---------------------------------------------------------------------------
// 6) implicit-GEMM 3x3 conv via WMMA with double-buffered async global->LDS
//    staging. One workgroup = 128 pixels x 128 out-channels; 8 waves, wave w
//    owns a 2(M)x4(N) block of 16x16 subtiles (shares A and B fragments:
//    12 ds_load_b128 per 8 WMMAs instead of 18).
template <bool RELU, typename OutT>
__global__ __launch_bounds__(256) void conv_kernel(const _Float16* __restrict__ in,   // [HW][128] fp16
                                                   const _Float16* __restrict__ wt,   // [9][128 o][128 c] fp16
                                                   const float* __restrict__ bias,    // [128]
                                                   OutT* __restrict__ out,            // [HW][128]
                                                   int H, int W) {
    const int HW = H * W;
    __shared__ __align__(32) _Float16 As[2][TILE_M * D];  // 2 x 32 KB
    __shared__ __align__(32) _Float16 Bs[2][D * D];       // 2 x 32 KB

    const int p0   = blockIdx.x * TILE_M;
    const int tid  = threadIdx.x;
    const int lane = tid & 31;
    const int wave = tid >> 5;
    const int l15  = lane & 15;
    const int half = lane >> 4;
    const int mg   = wave & 3;   // M rows  [mg*32, mg*32+32)
    const int ng   = wave >> 2;  // N cols  [ng*64, ng*64+64)

    v8f acc[2][4] = {};

    // async-stage one tap's A and B tiles into LDS buffer `buf`
    auto stage = [&](int tap, int buf) {
        const int dy = tap / 3 - 1;
        const int dx = tap % 3 - 1;
        // B tile: 2048 contiguous 16B chunks
        {
            const _Float16* src = wt + (size_t)tap * (D * D);
            #pragma unroll
            for (int j = 0; j < 8; ++j) {
                int chunk = tid + 256 * j;  // 0..2047
                uint32_t ldsoff = (uint32_t)(uintptr_t)(&Bs[buf][chunk * 8]);
                uint64_t ga     = (uint64_t)(uintptr_t)(src + chunk * 8);
                asm volatile("global_load_async_to_lds_b128 %0, %1, off"
                             :: "v"(ldsoff), "v"(ga) : "memory");
            }
        }
        // A tile: shifted pixel rows, zero-padded at grid borders
        #pragma unroll
        for (int j = 0; j < 8; ++j) {
            int chunk = tid + 256 * j;      // 0..2047
            int m  = chunk >> 4;            // pixel row in tile
            int kc = (chunk & 15) << 3;     // channel chunk
            int p  = p0 + m;
            bool valid = false;
            int q = 0;
            if (p < HW) {
                int y = p / W, x = p - y * W;
                int yy = y + dy, xx = x + dx;
                if ((unsigned)yy < (unsigned)H && (unsigned)xx < (unsigned)W) {
                    valid = true;
                    q = yy * W + xx;
                }
            }
            _Float16* dstp = &As[buf][m * D + kc];
            if (valid) {
                uint32_t ldsoff = (uint32_t)(uintptr_t)dstp;
                uint64_t ga     = (uint64_t)(uintptr_t)(in + (size_t)q * D + kc);
                asm volatile("global_load_async_to_lds_b128 %0, %1, off"
                             :: "v"(ldsoff), "v"(ga) : "memory");
            } else {
                v8h z;
                #pragma unroll
                for (int e = 0; e < 8; ++e) z[e] = (_Float16)0.0f;
                *(v8h*)dstp = z;
            }
        }
    };

    auto compute = [&](int buf) {
        #pragma unroll
        for (int k0 = 0; k0 < D; k0 += 32) {
            v16h a[2];
            #pragma unroll
            for (int im = 0; im < 2; ++im) {
                // A 16x32 layout (matches B):
                //   elems 0..15 : K = k0 + half*16 + 0..15
                const _Float16* arow = &As[buf][(mg * 32 + im * 16 + l15) * D];
                a[im] = *(const v16h*)(arow + k0 + half * 16);
            }
            #pragma unroll
            for (int jn = 0; jn < 4; ++jn) {
                // B fragment (32x16): K = k0 + half*16 + 0..15, contiguous per lane
                v16h b = *(const v16h*)(&Bs[buf][(ng * 64 + jn * 16 + l15) * D + k0 + half * 16]);
                #pragma unroll
                for (int im = 0; im < 2; ++im) {
                    acc[im][jn] = __builtin_amdgcn_wmma_f32_16x16x32_f16(
                        false, a[im], false, b, (short)0, acc[im][jn], false, false);
                }
            }
        }
    };

    // software pipeline: stage tap t+1 while computing tap t
    stage(0, 0);
    asm volatile("s_wait_asynccnt 0x0" ::: "memory");
    __syncthreads();
    for (int tap = 0; tap < 9; ++tap) {
        int buf = tap & 1;
        if (tap + 1 < 9) stage(tap + 1, buf ^ 1);
        compute(buf);
        asm volatile("s_wait_asynccnt 0x0" ::: "memory");
        __syncthreads();
    }

    // epilogue: C layout VGPR r -> M = r + 8*half, N = l15
    #pragma unroll
    for (int jn = 0; jn < 4; ++jn) {
        int n = ng * 64 + jn * 16 + l15;
        float bv = bias[n];
        #pragma unroll
        for (int im = 0; im < 2; ++im) {
            #pragma unroll
            for (int r = 0; r < 8; ++r) {
                int p = p0 + mg * 32 + im * 16 + r + half * 8;
                if (p < HW) {
                    float v = acc[im][jn][r] + bv;
                    if (RELU) v = v > 0.0f ? v : 0.0f;
                    out[(size_t)p * D + n] = (OutT)v;
                }
            }
        }
    }
}

// 7) gather + residual + layernorm; one wave32 per point, 4 channels per lane
__global__ __launch_bounds__(256) void gather_ln_kernel(const float* __restrict__ x2,
                                                        const float* __restrict__ feat,
                                                        const int* __restrict__ flat,
                                                        const float* __restrict__ lnw,
                                                        const float* __restrict__ lnb,
                                                        float* __restrict__ out, int N) {
    int wave = threadIdx.x >> 5;
    int lane = threadIdx.x & 31;
    int i = blockIdx.x * 8 + wave;
    if (i >= N) return;
    size_t gb = (size_t)flat[i] * D;
    size_t fb = (size_t)i * D;
    float v[4];
    float s = 0.0f;
    #pragma unroll
    for (int j = 0; j < 4; ++j) {
        int ch = lane + 32 * j;
        v[j] = x2[gb + ch] + feat[fb + ch];
        s += v[j];
    }
    #pragma unroll
    for (int o = 16; o >= 1; o >>= 1) s += __shfl_xor(s, o, 32);
    float mu = s * (1.0f / 128.0f);
    float d = 0.0f;
    #pragma unroll
    for (int j = 0; j < 4; ++j) { float t = v[j] - mu; d += t * t; }
    #pragma unroll
    for (int o = 16; o >= 1; o >>= 1) d += __shfl_xor(d, o, 32);
    float rs = rsqrtf(d * (1.0f / 128.0f) + 1e-5f);
    #pragma unroll
    for (int j = 0; j < 4; ++j) {
        int ch = lane + 32 * j;
        out[fb + ch] = (v[j] - mu) * rs * lnw[ch] + lnb[ch];
    }
}

// ---------------------------------------------------------------------------
extern "C" void kernel_launch(void* const* d_in, const int* in_sizes, int n_in,
                              void* d_out, int out_size, void* d_ws, size_t ws_size,
                              hipStream_t stream) {
    const float* features = (const float*)d_in[0];
    const float* coords   = (const float*)d_in[1];
    const float* w1       = (const float*)d_in[2];
    const float* b1       = (const float*)d_in[3];
    const float* w2       = (const float*)d_in[4];
    const float* b2       = (const float*)d_in[5];
    const float* lnw      = (const float*)d_in[6];
    const float* lnb      = (const float*)d_in[7];
    float* out = (float*)d_out;

    const int N = in_sizes[0] / D;                 // 131072
    const int H = (int)(sqrt((double)N) * 1.5);    // 543
    const int W = H;
    const size_t HW = (size_t)H * W;

    // workspace carve-up (256B aligned)
    uint8_t* base = (uint8_t*)d_ws;
    size_t off = 0;
    auto take = [&](size_t bytes) {
        off = (off + 255) & ~(size_t)255;
        size_t o = off;
        off += bytes;
        return o;
    };
    size_t sums_bytes = HW * D * sizeof(float) + HW * sizeof(float);  // sums + counts together
    float*     sums    = (float*)(base + take(sums_bytes));
    float*     counts  = sums + HW * D;
    float*     x2      = sums;  // reused after gridf16 is built
    _Float16*  gridf16 = (_Float16*)(base + take(HW * D * sizeof(_Float16)));
    _Float16*  x1      = (_Float16*)(base + take(HW * D * sizeof(_Float16)));
    _Float16*  w1h     = (_Float16*)(base + take((size_t)9 * D * D * sizeof(_Float16)));
    _Float16*  w2h     = (_Float16*)(base + take((size_t)9 * D * D * sizeof(_Float16)));
    int*       mm      = (int*)(base + take(16));
    int*       flat    = (int*)(base + take((size_t)N * sizeof(int)));

    hipMemsetAsync(sums, 0, sums_bytes, stream);
    init_mm_kernel<<<1, 1, 0, stream>>>(mm);
    minmax_kernel<<<(2 * N + 255) / 256, 256, 0, stream>>>(coords, mm, 2 * N);
    scatter_kernel<<<N, 128, 0, stream>>>(features, coords, mm, flat, sums, counts, H, W);
    const int wtot = 9 * D * D;
    wprep_kernel<<<(2 * wtot + 255) / 256, 256, 0, stream>>>(w1, w2, w1h, w2h, wtot);
    grid_norm_kernel<<<(int)((HW * D + 255) / 256), 256, 0, stream>>>(sums, counts, gridf16, HW * D);
    const int cblocks = (int)((HW + TILE_M - 1) / TILE_M);
    conv_kernel<true,  _Float16><<<cblocks, 256, 0, stream>>>(gridf16, w1h, b1, x1, H, W);
    conv_kernel<false, float   ><<<cblocks, 256, 0, stream>>>(x1,      w2h, b2, x2, H, W);
    gather_ln_kernel<<<(N + 7) / 8, 256, 0, stream>>>(x2, features, flat, lnw, lnb, out, N);
}
